// HeteroGraphormerLayerComplete_72456098283985
// MI455X (gfx1250) — matrix operand order repaired and checked by reading, not compiled, not run on hardware
//
#include <hip/hip_runtime.h>
#include <hip/hip_bf16.h>
#include <math.h>

typedef __bf16 bf16;
typedef __attribute__((ext_vector_type(16))) __bf16 v16bf;
typedef __attribute__((ext_vector_type(8)))  float  v8f;

#define NNODE 4096
#define CDIM  128
#define NHEAD 4
#define HDIM  32
#define NEDGE 65536

// ---- WMMA fragment index maps (ISA 7.12.2, wave32) ----
// A (16x32 bf16): lane holds row M=lane&15; element e -> K (two 8-elem runs):
__device__ __forceinline__ int kmapA(int e, int half) {
  return ((e >> 3) << 4) + (half << 3) + (e & 7);
}
// B (32x16 bf16): lane holds col N=lane&15; element e -> K (two 8-elem runs):
__device__ __forceinline__ int kmapB(int e, int half) {
  return ((e >> 3) << 4) + (half << 3) + (e & 7);
}

__device__ __forceinline__ v8f wmma_bf16(v16bf a, v16bf b, v8f c) {
  return __builtin_amdgcn_wmma_f32_16x16x32_bf16(false, a, false, b, (short)0, c,
                                                 false, false);
}

// ---------------- float -> bf16, transposed (for 128x128 weights) ----------
// dst[n*128 + k] = src[k*128 + n]  -> B-fragment reads become contiguous.
__global__ void f2bfT_kernel(const float* __restrict__ src, bf16* __restrict__ dst) {
  int i = blockIdx.x * blockDim.x + threadIdx.x;  // 16384 threads
  int k = i >> 7, n = i & 127;
  dst[n * CDIM + k] = (bf16)src[k * CDIM + n];
}

// ---------------- LayerNorm (one row of 128 per block, 128 threads) -------
__global__ void ln_kernel(const float* __restrict__ x, const float* __restrict__ g,
                          const float* __restrict__ b, float* __restrict__ outF,
                          bf16* __restrict__ outB) {
  int row = blockIdx.x;
  int tid = threadIdx.x;  // 0..127
  float v = x[row * CDIM + tid];
  float s = v, s2 = v * v;
#pragma unroll
  for (int off = 16; off; off >>= 1) {
    s += __shfl_down(s, off, 32);
    s2 += __shfl_down(s2, off, 32);
  }
  __shared__ float red[8];
  int wave = tid >> 5, lane = tid & 31;
  if (lane == 0) { red[wave] = s; red[4 + wave] = s2; }
  __syncthreads();
  float ts = red[0] + red[1] + red[2] + red[3];
  float ts2 = red[4] + red[5] + red[6] + red[7];
  float mean = ts * (1.0f / CDIM);
  float var = ts2 * (1.0f / CDIM) - mean * mean;
  float rstd = rsqrtf(var + 1e-5f);
  float y = (v - mean) * rstd * g[tid] + b[tid];
  if (outF) outF[row * CDIM + tid] = y;
  if (outB) outB[row * CDIM + tid] = (bf16)y;
}

// ---------------- Degree (segment-sum of ones via atomics) ----------------
__global__ void zero_kernel(float* __restrict__ p, int n) {
  int i = blockIdx.x * blockDim.x + threadIdx.x;
  if (i < n) p[i] = 0.0f;
}

__global__ void degree_kernel(const int* __restrict__ eab, const int* __restrict__ eba,
                              float* __restrict__ degA, float* __restrict__ degB) {
  int i = blockIdx.x * blockDim.x + threadIdx.x;
  if (i < NEDGE) {
    atomicAdd(&degA[eab[i]], 1.0f);          // edge_ab row0: src in A
    atomicAdd(&degB[eab[NEDGE + i]], 1.0f);  // edge_ab row1: dst in B
    atomicAdd(&degB[eba[i]], 1.0f);          // edge_ba row0: src in B
    atomicAdd(&degA[eba[NEDGE + i]], 1.0f);  // edge_ba row1: dst in A
  }
}

// ---------------- Generic 128-wide WMMA GEMM:  Y = A[Mx128] @ W ------------
// W is stored TRANSPOSED: Wt[n*128 + k].  block = 256 threads (8 waves), each
// wave owns one 16-col N tile; block owns 16 rows.
// mode: 0 = bf16 out (Q/K proj), 1 = f32 out + bias + deg + resid,
// 2 = gelu -> bf16 out, 3 = f32 out + bias + resid,
// 4 = bf16 out transposed [col*NNODE + row] (V proj for flash P@V).
__global__ void gemm128_kernel(const bf16* __restrict__ A, const bf16* __restrict__ Wt,
                               const float* __restrict__ bias, int mode,
                               const float* __restrict__ resid,
                               const float* __restrict__ deg,
                               float* __restrict__ outF, bf16* __restrict__ outB) {
  int lane = threadIdx.x & 31;
  int wave = threadIdx.x >> 5;
  int m = lane & 15, half = lane >> 4;
  int mbase = blockIdx.x * 16;
  int nbase = wave * 16;

  v8f acc = {};
#pragma unroll
  for (int kb = 0; kb < CDIM; kb += 32) {
    v16bf af, bf;
#pragma unroll
    for (int e = 0; e < 16; ++e) {
      af[e] = A[(mbase + m) * CDIM + kb + kmapA(e, half)];
      bf[e] = Wt[(nbase + m) * CDIM + kb + kmapB(e, half)];  // contiguous run
    }
    acc = wmma_bf16(af, bf, acc);
  }

#pragma unroll
  for (int r = 0; r < 8; ++r) {
    int row = mbase + r + 8 * half;
    int col = nbase + m;
    float y = acc[r] + bias[col];
    if (mode == 0) {
      outB[row * CDIM + col] = (bf16)y;
    } else if (mode == 1) {
      outF[row * CDIM + col] = y + deg[row] + resid[row * CDIM + col];
    } else if (mode == 2) {
      float t = 0.5f * y * (1.0f + erff(y * 0.70710678118654752f));
      outB[row * CDIM + col] = (bf16)t;
    } else if (mode == 3) {
      outF[row * CDIM + col] = y + resid[row * CDIM + col];
    } else {  // mode 4: transposed store, contiguous in r -> packs to b128
      outB[col * NNODE + row] = (bf16)y;
    }
  }
}

// ---------------- Flash cross-attention ------------------------------------
// grid = (nQ/64, H), block = 128 (4 waves).  Each wave: one 16-query tile of
// one head; streams all keys in chunks of 32 with online softmax.
// Q, K row-major [node][128] bf16; V transposed Vt[d][node] bf16.
__global__ void flash_attn_kernel(const bf16* __restrict__ Q, const bf16* __restrict__ K,
                                  const bf16* __restrict__ Vt,
                                  const float* __restrict__ relp,
                                  bf16* __restrict__ outB, int nK) {
  int lane = threadIdx.x & 31;
  int wave = threadIdx.x >> 5;
  int m = lane & 15, half = lane >> 4;
  int qbase = (blockIdx.x * 4 + wave) * 16;
  int hoff = blockIdx.y * HDIM;
  float rel = relp[0];
  const float scl = 0.17677669529663687f;  // 1/sqrt(32)

  v16bf aq;
#pragma unroll
  for (int e = 0; e < 16; ++e)
    aq[e] = Q[(qbase + m) * CDIM + hoff + kmapA(e, half)];

  v8f o0 = {}, o1 = {};
  float rm[8], rl[8];
#pragma unroll
  for (int r = 0; r < 8; ++r) { rm[r] = -1e30f; rl[r] = 0.0f; }

  __shared__ float p_sm[4][16][32];  // per-wave P staging (C-layout -> A-layout)

  for (int kc = 0; kc < nK; kc += 32) {
    // prefetch next chunk (global_prefetch_b8 path)
    if (kc + 32 < nK) {
      __builtin_prefetch(&K[(kc + 32 + m + 16 * half) * CDIM + hoff], 0, 1);
      __builtin_prefetch(&Vt[(hoff + m) * nK + kc + 32 + 16 * half], 0, 1);
    }

    // S = Q @ K^T for 32 keys (two 16-key column tiles); contiguous K reads
    v16bf bk0, bk1;
#pragma unroll
    for (int e = 0; e < 16; ++e) {
      int kk = hoff + kmapB(e, half);
      bk0[e] = K[(kc + m) * CDIM + kk];
      bk1[e] = K[(kc + 16 + m) * CDIM + kk];
    }
    v8f s0 = {}, s1 = {};
    s0 = wmma_bf16(aq, bk0, s0);
    s1 = wmma_bf16(aq, bk1, s1);

    // online softmax update (per-row stats; row = r + 8*half, cols across lanes)
#pragma unroll
    for (int r = 0; r < 8; ++r) {
      float v0 = s0[r] * scl + rel;
      float v1 = s1[r] * scl + rel;
      float cm = fmaxf(v0, v1);
#pragma unroll
      for (int off = 8; off; off >>= 1) cm = fmaxf(cm, __shfl_xor(cm, off, 16));
      float nm = fmaxf(rm[r], cm);
      float sc = __expf(rm[r] - nm);
      float p0 = __expf(v0 - nm);
      float p1 = __expf(v1 - nm);
      float rs = p0 + p1;
#pragma unroll
      for (int off = 8; off; off >>= 1) rs += __shfl_xor(rs, off, 16);
      rl[r] = rl[r] * sc + rs;
      rm[r] = nm;
      o0[r] *= sc;
      o1[r] *= sc;
      int prow = r + 8 * half;
      p_sm[wave][prow][m] = p0;
      p_sm[wave][prow][16 + m] = p1;
    }
    // wave-private LDS RAW: drain DS counter before re-reading as A fragment
    asm volatile("s_wait_dscnt 0" ::: "memory");

    // O += P @ V   (P: 16x32; Vt gives two contiguous 16B runs per lane)
    v16bf ap, bv0, bv1;
#pragma unroll
    for (int e = 0; e < 16; ++e) {
      ap[e] = (bf16)p_sm[wave][m][kmapA(e, half)];
      int kk = kc + kmapB(e, half);
      bv0[e] = Vt[(hoff + m) * nK + kk];
      bv1[e] = Vt[(hoff + 16 + m) * nK + kk];
    }
    o0 = wmma_bf16(ap, bv0, o0);
    o1 = wmma_bf16(ap, bv1, o1);
  }

#pragma unroll
  for (int r = 0; r < 8; ++r) {
    float inv = 1.0f / rl[r];
    int row = qbase + r + 8 * half;
    outB[row * CDIM + hoff + m] = (bf16)(o0[r] * inv);
    outB[row * CDIM + hoff + 16 + m] = (bf16)(o1[r] * inv);
  }
}

// ---------------------------------------------------------------------------
extern "C" void kernel_launch(void* const* d_in, const int* in_sizes, int n_in,
                              void* d_out, int out_size, void* d_ws, size_t ws_size,
                              hipStream_t stream) {
  (void)in_sizes; (void)n_in; (void)out_size; (void)ws_size;
  const float* x_a = (const float*)d_in[0];
  const float* x_b = (const float*)d_in[1];
  const int* eab = (const int*)d_in[2];
  const int* eba = (const int*)d_in[3];
  const float* Wq = (const float*)d_in[4];  const float* bq = (const float*)d_in[5];
  const float* Wk = (const float*)d_in[6];  const float* bk = (const float*)d_in[7];
  const float* Wv = (const float*)d_in[8];  const float* bv = (const float*)d_in[9];
  const float* Wo = (const float*)d_in[10]; const float* bo = (const float*)d_in[11];
  const float* g_attn = (const float*)d_in[12]; const float* b_attn = (const float*)d_in[13];
  const float* g_ffn = (const float*)d_in[14];  const float* b_ffn = (const float*)d_in[15];
  const float* g_f = (const float*)d_in[16];    const float* b_f = (const float*)d_in[17];
  const float* W1 = (const float*)d_in[18]; const float* b1 = (const float*)d_in[19];
  const float* W2 = (const float*)d_in[20]; const float* b2 = (const float*)d_in[21];
  const float* rel_ab = (const float*)d_in[22];
  const float* rel_ba = (const float*)d_in[23];
  float* out = (float*)d_out;

  // ---- workspace carve-up ----
  char* wsc = (char*)d_ws;
  size_t off = 0;
  auto alloc = [&](size_t bytes) -> void* {
    void* p = wsc + off;
    off += (bytes + 255) & ~(size_t)255;
    return p;
  };
  const size_t NC = (size_t)NNODE * CDIM;
  const size_t WB = (size_t)CDIM * CDIM * sizeof(bf16);
  bf16* Wqb = (bf16*)alloc(WB); bf16* Wkb = (bf16*)alloc(WB);
  bf16* Wvb = (bf16*)alloc(WB); bf16* Wob = (bf16*)alloc(WB);
  bf16* W1b = (bf16*)alloc(WB); bf16* W2b = (bf16*)alloc(WB);
  bf16* xnb_a = (bf16*)alloc(NC * 2); bf16* xnb_b = (bf16*)alloc(NC * 2);
  bf16* Qa = (bf16*)alloc(NC * 2); bf16* Ka = (bf16*)alloc(NC * 2); bf16* Va = (bf16*)alloc(NC * 2);
  bf16* Qb = (bf16*)alloc(NC * 2); bf16* Kb = (bf16*)alloc(NC * 2); bf16* Vb = (bf16*)alloc(NC * 2);
  bf16* attA = (bf16*)alloc(NC * 2); bf16* attB = (bf16*)alloc(NC * 2);
  bf16* ffnA = (bf16*)alloc(NC * 2); bf16* gelA = (bf16*)alloc(NC * 2);
  bf16* ffnB = (bf16*)alloc(NC * 2); bf16* gelB = (bf16*)alloc(NC * 2);
  float* resA = (float*)alloc(NC * 4); float* resB = (float*)alloc(NC * 4);
  float* tmpF = (float*)alloc(NC * 4);
  float* degA = (float*)alloc(NNODE * 4); float* degB = (float*)alloc(NNODE * 4);

  // weights -> bf16, TRANSPOSED (Wt[n][k])
  const int WN = CDIM * CDIM;  // 16384
  dim3 cvt((WN + 255) / 256), thr(256);
  f2bfT_kernel<<<cvt, thr, 0, stream>>>(Wq, Wqb);
  f2bfT_kernel<<<cvt, thr, 0, stream>>>(Wk, Wkb);
  f2bfT_kernel<<<cvt, thr, 0, stream>>>(Wv, Wvb);
  f2bfT_kernel<<<cvt, thr, 0, stream>>>(Wo, Wob);
  f2bfT_kernel<<<cvt, thr, 0, stream>>>(W1, W1b);
  f2bfT_kernel<<<cvt, thr, 0, stream>>>(W2, W2b);

  // pre-attention LN -> bf16
  ln_kernel<<<NNODE, 128, 0, stream>>>(x_a, g_attn, b_attn, nullptr, xnb_a);
  ln_kernel<<<NNODE, 128, 0, stream>>>(x_b, g_attn, b_attn, nullptr, xnb_b);

  // Q/K projections (mode 0, row-major out); V projection (mode 4, transposed)
  dim3 ggrid(NNODE / 16), gblk(256);
  gemm128_kernel<<<ggrid, gblk, 0, stream>>>(xnb_a, Wqb, bq, 0, nullptr, nullptr, nullptr, Qa);
  gemm128_kernel<<<ggrid, gblk, 0, stream>>>(xnb_a, Wkb, bk, 0, nullptr, nullptr, nullptr, Ka);
  gemm128_kernel<<<ggrid, gblk, 0, stream>>>(xnb_a, Wvb, bv, 4, nullptr, nullptr, nullptr, Va);
  gemm128_kernel<<<ggrid, gblk, 0, stream>>>(xnb_b, Wqb, bq, 0, nullptr, nullptr, nullptr, Qb);
  gemm128_kernel<<<ggrid, gblk, 0, stream>>>(xnb_b, Wkb, bk, 0, nullptr, nullptr, nullptr, Kb);
  gemm128_kernel<<<ggrid, gblk, 0, stream>>>(xnb_b, Wvb, bv, 4, nullptr, nullptr, nullptr, Vb);

  // degrees
  zero_kernel<<<(NNODE + 255) / 256, 256, 0, stream>>>(degA, NNODE);
  zero_kernel<<<(NNODE + 255) / 256, 256, 0, stream>>>(degB, NNODE);
  degree_kernel<<<(NEDGE + 255) / 256, 256, 0, stream>>>(eab, eba, degA, degB);

  // cross attention (flash):  A queries over B keys with rel_ba, and vice versa
  dim3 fgrid(NNODE / 64, NHEAD), fblk(128);
  flash_attn_kernel<<<fgrid, fblk, 0, stream>>>(Qa, Kb, Vb, rel_ba, attA, NNODE);
  flash_attn_kernel<<<fgrid, fblk, 0, stream>>>(Qb, Ka, Va, rel_ab, attB, NNODE);

  // output projection + degree + residual (mode 1)
  gemm128_kernel<<<ggrid, gblk, 0, stream>>>(attA, Wob, bo, 1, x_a, degA, resA, nullptr);
  gemm128_kernel<<<ggrid, gblk, 0, stream>>>(attB, Wob, bo, 1, x_b, degB, resB, nullptr);

  // FFN for A: LN(LN(res)) -> gelu(W1) -> W2 + residual -> out[0]
  ln_kernel<<<NNODE, 128, 0, stream>>>(resA, g_ffn, b_ffn, tmpF, nullptr);
  ln_kernel<<<NNODE, 128, 0, stream>>>(tmpF, g_f, b_f, nullptr, ffnA);
  gemm128_kernel<<<ggrid, gblk, 0, stream>>>(ffnA, W1b, b1, 2, nullptr, nullptr, nullptr, gelA);
  gemm128_kernel<<<ggrid, gblk, 0, stream>>>(gelA, W2b, b2, 3, resA, nullptr, out, nullptr);

  // FFN for B -> out[1]
  ln_kernel<<<NNODE, 128, 0, stream>>>(resB, g_ffn, b_ffn, tmpF, nullptr);
  ln_kernel<<<NNODE, 128, 0, stream>>>(tmpF, g_f, b_f, nullptr, ffnB);
  gemm128_kernel<<<ggrid, gblk, 0, stream>>>(ffnB, W1b, b1, 2, nullptr, nullptr, nullptr, gelB);
  gemm128_kernel<<<ggrid, gblk, 0, stream>>>(gelB, W2b, b2, 3, resB, nullptr, out + NC, nullptr);
}
